// VoteNet_80917183857083
// MI455X (gfx1250) — compile-verified, matrix operand order checked by reading, not run on hardware
//
#include <hip/hip_runtime.h>

// ---------------------------------------------------------------------------
// PointNet++ (VoteNet backbone) for MI455X / gfx1250.
//  * MLP GEMMs via v_wmma_f32_16x16x32_f16 (f16 A/B, f32 accum), K padded to 32.
//    Each wave computes a 32x64 output strip (2 M-tiles x 4 N-tiles, 8 accums)
//    so 8 WMMAs share 12 b128 loads -> ~2.7x better L2->reg intensity.
//  * FPS per batch in one workgroup with the full point cloud resident in the
//    320KB WGP LDS (xyz + min-dist = 256KB for N=16384).
// ---------------------------------------------------------------------------

typedef __attribute__((ext_vector_type(16))) _Float16 v16h;
typedef __attribute__((ext_vector_type(8)))  _Float16 v8h;
typedef __attribute__((ext_vector_type(8)))  float    v8f;

#define B_  2
#define N0_ 16384

// ---------------- transpose (B,3,N) -> (B,N,3) -----------------------------
__global__ void k_transpose(const float* __restrict__ pc, float* __restrict__ xyz,
                            int N, int total) {
    int i = blockIdx.x * blockDim.x + threadIdx.x;
    if (i >= total) return;                 // total = B*N
    int b = i / N, n = i % N;
    const float* base = pc + (size_t)b * 3 * N;
    xyz[(size_t)i * 3 + 0] = base[0 * N + n];
    xyz[(size_t)i * 3 + 1] = base[1 * N + n];
    xyz[(size_t)i * 3 + 2] = base[2 * N + n];
}

// ---------------- farthest point sampling (LDS-resident) -------------------
extern __shared__ float fps_smem[];
__global__ void k_fps(const float* __restrict__ xyz, int N, int np,
                      int* __restrict__ inds) {
    const int b  = blockIdx.x;
    const int t  = threadIdx.x;
    const int nt = blockDim.x;
    float* sx = fps_smem;
    float* sy = sx + N;
    float* sz = sy + N;
    float* md = sz + N;
    float* rv = md + N;
    int*   ri = (int*)(rv + nt);
    const float* base = xyz + (size_t)b * N * 3;
    for (int i = t; i < N; i += nt) {
        sx[i] = base[(size_t)i * 3 + 0];
        sy[i] = base[(size_t)i * 3 + 1];
        sz[i] = base[(size_t)i * 3 + 2];
        md[i] = 1e10f;
    }
    __syncthreads();
    int far = 0;
    for (int it = 0; it < np; ++it) {
        if (t == 0) inds[b * np + it] = far;
        float cx = sx[far], cy = sy[far], cz = sz[far];
        float bestv = -1.0f; int besti = 0;
        for (int i = t; i < N; i += nt) {
            float dx = sx[i] - cx, dy = sy[i] - cy, dz = sz[i] - cz;
            float d  = dx * dx + dy * dy + dz * dz;
            float m  = fminf(md[i], d);
            md[i] = m;
            if (m > bestv) { bestv = m; besti = i; }
        }
        rv[t] = bestv; ri[t] = besti;
        __syncthreads();
        for (int s = nt >> 1; s > 0; s >>= 1) {
            if (t < s) {
                if (rv[t + s] > rv[t] ||
                    (rv[t + s] == rv[t] && ri[t + s] < ri[t])) {
                    rv[t] = rv[t + s]; ri[t] = ri[t + s];
                }
            }
            __syncthreads();
        }
        far = ri[0];
        __syncthreads();
    }
}

// ---------------- gather sampled centers ------------------------------------
__global__ void k_gather_xyz(const float* __restrict__ xyz, const int* __restrict__ inds,
                             float* __restrict__ out, int Nin, int np, int total) {
    int i = blockIdx.x * blockDim.x + threadIdx.x;
    if (i >= total) return;                 // total = B*np
    int b = i / np;
    int p = inds[i];
    const float* src = xyz + ((size_t)b * Nin + p) * 3;
    out[(size_t)i * 3 + 0] = src[0];
    out[(size_t)i * 3 + 1] = src[1];
    out[(size_t)i * 3 + 2] = src[2];
}

// ---------------- ball query (first nsample in-radius, ascending) -----------
__global__ void k_ball_query(const float* __restrict__ xyz, const float* __restrict__ nxyz,
                             int N, int np, int ns, float r2,
                             int* __restrict__ idx, int total) {
    int i = blockIdx.x * blockDim.x + threadIdx.x;
    if (i >= total) return;                 // total = B*np
    int b = i / np;
    const float* q    = nxyz + (size_t)i * 3;
    const float* base = xyz + (size_t)b * N * 3;
    int* out = idx + (size_t)i * ns;
    int cnt = 0, first = 0;
    float qx = q[0], qy = q[1], qz = q[2];
    for (int j = 0; j < N && cnt < ns; ++j) {
        float dx = base[(size_t)j * 3 + 0] - qx;
        float dy = base[(size_t)j * 3 + 1] - qy;
        float dz = base[(size_t)j * 3 + 2] - qz;
        float d = dx * dx + dy * dy + dz * dz;
        if (d < r2) {
            if (cnt == 0) first = j;
            out[cnt++] = j;
        }
    }
    for (; cnt < ns; ++cnt) out[cnt] = first;
}

// ---------------- grouping -> padded f16 A matrix ---------------------------
__global__ void k_group(const float* __restrict__ xyz, const float* __restrict__ nxyz,
                        const float* __restrict__ feats, const int* __restrict__ idx,
                        _Float16* __restrict__ A, int np, int ns, int Cf, int Cpad,
                        float inv_r, int Nin, size_t total) {
    size_t i = (size_t)blockIdx.x * blockDim.x + threadIdx.x;
    if (i >= total) return;                 // total = B*np*ns*Cpad
    int    c  = (int)(i % Cpad);
    size_t r  = i / Cpad;                   // grouped row (b,s,j)
    size_t bs = r / ns;                     // (b,s)
    int    b  = (int)(bs / np);
    int    p  = idx[r];
    float v = 0.0f;
    if (c < 3) {
        v = (xyz[((size_t)b * Nin + p) * 3 + c] - nxyz[bs * 3 + c]) * inv_r;
    } else if (c - 3 < Cf) {
        v = feats[((size_t)b * Nin + p) * Cf + (c - 3)];
    }
    A[i] = (_Float16)v;
}

// ---------------- weight convert + K-pad to f16 -----------------------------
__global__ void k_wconv(const float* __restrict__ W, _Float16* __restrict__ Wp,
                        int Ci, int Kp, int total) {
    int i = blockIdx.x * blockDim.x + threadIdx.x;
    if (i >= total) return;                 // total = Co*Kp
    int c = i % Kp, o = i / Kp;
    Wp[i] = (c < Ci) ? (_Float16)W[(size_t)o * Ci + c] : (_Float16)0.0f;
}

// ---------------- WMMA GEMM: O[M,Co] = A[M,Kp] * W[Co,Kp]^T -----------------
// Each wave computes a 32x64 strip: 2 M-tiles x 4 N-tiles = 8 accumulators.
#define GEMM_WAVES 4
__global__ void k_gemm_wmma(const _Float16* __restrict__ A,
                            const _Float16* __restrict__ W,
                            _Float16* __restrict__ O,
                            int M, int Kp, int Co) {
    const int lane = threadIdx.x & 31;
    const int wave = threadIdx.x >> 5;
    const int m0   = (blockIdx.y * GEMM_WAVES + wave) * 32;   // wave-uniform
    if (m0 >= M) return;                                      // uniform per wave
    const int n0 = blockIdx.x * 64;
    const int lm = lane & 15;
    const int hi = lane >> 4;
    const _Float16* ap0 = A + (size_t)(m0 + lm) * Kp + hi * 8;
    const _Float16* ap1 = ap0 + (size_t)16 * Kp;
    const _Float16* bp  = W + (size_t)(n0 + lm) * Kp + hi * 8;
    const size_t bstride = (size_t)16 * Kp;

    v8f acc[8];
#pragma unroll
    for (int q = 0; q < 8; ++q)
        acc[q] = (v8f){0.f, 0.f, 0.f, 0.f, 0.f, 0.f, 0.f, 0.f};

    for (int k = 0; k < Kp; k += 32) {
        v16h a0, a1, bf[4];
        {
            v8h lo0 = *(const v8h*)(ap0 + k);
            v8h hi0 = *(const v8h*)(ap0 + k + 16);
            v8h lo1 = *(const v8h*)(ap1 + k);
            v8h hi1 = *(const v8h*)(ap1 + k + 16);
#pragma unroll
            for (int q = 0; q < 8; ++q) {
                a0[q] = lo0[q]; a0[q + 8] = hi0[q];
                a1[q] = lo1[q]; a1[q + 8] = hi1[q];
            }
        }
#pragma unroll
        for (int j = 0; j < 4; ++j) {
            v8h lo = *(const v8h*)(bp + j * bstride + k);
            v8h hh = *(const v8h*)(bp + j * bstride + k + 16);
#pragma unroll
            for (int q = 0; q < 8; ++q) { bf[j][q] = lo[q]; bf[j][q + 8] = hh[q]; }
        }
#pragma unroll
        for (int j = 0; j < 4; ++j) {
            acc[j]     = __builtin_amdgcn_wmma_f32_16x16x32_f16(
                false, a0, false, bf[j], (short)0, acc[j],     false, false);
            acc[4 + j] = __builtin_amdgcn_wmma_f32_16x16x32_f16(
                false, a1, false, bf[j], (short)0, acc[4 + j], false, false);
        }
    }

#pragma unroll
    for (int mi = 0; mi < 2; ++mi) {
#pragma unroll
        for (int j = 0; j < 4; ++j) {
            _Float16* op = O + (size_t)(m0 + mi * 16) * Co + n0 + j * 16 + lm;
            v8f a = acc[mi * 4 + j];
#pragma unroll
            for (int r = 0; r < 8; ++r)
                op[(size_t)(r + hi * 8) * Co] = (_Float16)a[r];
        }
    }
}

// ---------------- per-channel sum / sumsq -----------------------------------
__global__ void k_col_stats(const _Float16* __restrict__ X, int M, int C,
                            float* __restrict__ sums, float* __restrict__ sqs) {
    __shared__ float s1[256], s2[256];
    int c = blockIdx.x, t = threadIdx.x;
    float a = 0.f, b = 0.f;
    for (int m = t; m < M; m += blockDim.x) {
        float v = (float)X[(size_t)m * C + c];
        a += v; b += v * v;
    }
    s1[t] = a; s2[t] = b;
    __syncthreads();
    for (int s = 128; s > 0; s >>= 1) {
        if (t < s) { s1[t] += s1[t + s]; s2[t] += s2[t + s]; }
        __syncthreads();
    }
    if (t == 0) { sums[c] = s1[0]; sqs[c] = s2[0]; }
}

// ---------------- batch-norm (batch stats) + ReLU + f16 repack --------------
__global__ void k_bn_relu(const _Float16* __restrict__ X,
                          const float* __restrict__ g, const float* __restrict__ bta,
                          const float* __restrict__ sums, const float* __restrict__ sqs,
                          int M, int C, _Float16* __restrict__ O16,
                          float* __restrict__ O32) {
    size_t i = (size_t)blockIdx.x * blockDim.x + threadIdx.x;
    size_t tot = (size_t)M * C;
    if (i >= tot) return;
    int c = (int)(i % C);
    float mean = sums[c] / (float)M;
    float var  = sqs[c] / (float)M - mean * mean;
    float y = g[c] * ((float)X[i] - mean) * rsqrtf(var + 1e-5f) + bta[c];
    y = fmaxf(y, 0.0f);
    O16[i] = (_Float16)y;
    if (O32) O32[i] = y;
}

// ---------------- max over the nsample axis ---------------------------------
__global__ void k_max_pool(const _Float16* __restrict__ gin, float* __restrict__ out,
                           int ns, int C, int total) {
    int i = blockIdx.x * blockDim.x + threadIdx.x;
    if (i >= total) return;                 // total = B*np*C
    int    c  = i % C;
    size_t bs = (size_t)i / C;
    float m = -1e30f;
    for (int j = 0; j < ns; ++j)
        m = fmaxf(m, (float)gin[(bs * ns + j) * C + c]);
    out[i] = m;
}

// ---------------- 3-NN interpolation + concat -> f16 A ----------------------
__global__ void k_three_interp(const float* __restrict__ xyz1, const float* __restrict__ xyz2,
                               const float* __restrict__ f1, const float* __restrict__ f2,
                               _Float16* __restrict__ A, int N1, int N2,
                               int C1, int C2, int total) {
    int i = blockIdx.x * blockDim.x + threadIdx.x;
    if (i >= total) return;                 // total = B*N1
    int b = i / N1;
    const float* q    = xyz1 + (size_t)i * 3;
    const float* base = xyz2 + (size_t)b * N2 * 3;
    float qx = q[0], qy = q[1], qz = q[2];
    float d0 = 1e30f, d1 = 1e30f, d2 = 1e30f;
    int   i0 = 0, i1 = 0, i2 = 0;
    for (int j = 0; j < N2; ++j) {
        float dx = base[(size_t)j * 3 + 0] - qx;
        float dy = base[(size_t)j * 3 + 1] - qy;
        float dz = base[(size_t)j * 3 + 2] - qz;
        float d = dx * dx + dy * dy + dz * dz;
        if (d < d0)      { d2 = d1; i2 = i1; d1 = d0; i1 = i0; d0 = d; i0 = j; }
        else if (d < d1) { d2 = d1; i2 = i1; d1 = d;  i1 = j; }
        else if (d < d2) { d2 = d;  i2 = j; }
    }
    float w0 = 1.f / (d0 + 1e-8f), w1 = 1.f / (d1 + 1e-8f), w2 = 1.f / (d2 + 1e-8f);
    float ws = w0 + w1 + w2;
    w0 /= ws; w1 /= ws; w2 /= ws;
    _Float16* row = A + (size_t)i * (C1 + C2);
    const float* f1r = f1 + (size_t)i * C1;
    const float* p0 = f2 + ((size_t)b * N2 + i0) * C2;
    const float* p1 = f2 + ((size_t)b * N2 + i1) * C2;
    const float* p2 = f2 + ((size_t)b * N2 + i2) * C2;
    for (int c = 0; c < C1; ++c) row[c] = (_Float16)f1r[c];
    for (int c = 0; c < C2; ++c)
        row[C1 + c] = (_Float16)(w0 * p0[c] + w1 * p1[c] + w2 * p2[c]);
}

// ---------------- channel-wise max (cls) ------------------------------------
__global__ void k_ch_max(const _Float16* __restrict__ gin, float* __restrict__ cls,
                         int C, int total) {
    int i = blockIdx.x * blockDim.x + threadIdx.x;
    if (i >= total) return;                 // total = B*1024
    const _Float16* r = gin + (size_t)i * C;
    float m = -1e30f;
    for (int c = 0; c < C; ++c) m = fmaxf(m, (float)r[c]);
    cls[i] = m;
}

// ---------------- final linear ---------------------------------------------
__global__ void k_linear(const float* __restrict__ cls, const float* __restrict__ W,
                         float* __restrict__ out, int Nn, int E, int total) {
    int i = blockIdx.x * blockDim.x + threadIdx.x;
    if (i >= total) return;                 // total = B*E
    int b = i / E, e = i % E;
    const float* cr = cls + (size_t)b * Nn;
    const float* wr = W + (size_t)e * Nn;
    float acc = 0.f;
    for (int n = 0; n < Nn; ++n) acc += cr[n] * wr[n];
    out[i] = acc;
}

// ---------------- final batch-norm over B=2 ---------------------------------
__global__ void k_final_bn(const float* __restrict__ o, const float* __restrict__ g,
                           const float* __restrict__ b, float* __restrict__ out, int E) {
    int e = blockIdx.x * blockDim.x + threadIdx.x;
    if (e >= E) return;
    float x0 = o[e], x1 = o[E + e];
    float m = 0.5f * (x0 + x1);
    float v = 0.5f * ((x0 - m) * (x0 - m) + (x1 - m) * (x1 - m));
    float s = rsqrtf(v + 1e-5f);
    out[e]     = g[e] * (x0 - m) * s + b[e];
    out[E + e] = g[e] * (x1 - m) * s + b[e];
}

// ===========================================================================
extern "C" void kernel_launch(void* const* d_in, const int* in_sizes, int n_in,
                              void* d_out, int out_size, void* d_ws, size_t ws_size,
                              hipStream_t stream) {
    (void)in_sizes; (void)n_in; (void)out_size; (void)ws_size;

    // ---- workspace carving ----
    char* p = (char*)d_ws;
    auto carve = [&](size_t bytes) -> void* {
        void* r = (void*)p;
        p += (bytes + 255) & ~(size_t)255;
        return r;
    };
    float* xyz0 = (float*)carve((size_t)B_ * N0_ * 3 * 4);
    float* xyz1 = (float*)carve((size_t)B_ * 2048 * 3 * 4);
    float* xyz2 = (float*)carve((size_t)B_ * 1024 * 3 * 4);
    float* xyz3 = (float*)carve((size_t)B_ * 512  * 3 * 4);
    float* xyz4 = (float*)carve((size_t)B_ * 256  * 3 * 4);
    int*   inds = (int*)  carve((size_t)B_ * 2048 * 4);
    int*   idx  = (int*)  carve((size_t)B_ * 2048 * 64 * 4);
    float* f1   = (float*)carve((size_t)B_ * 2048 * 128 * 4);
    float* f2   = (float*)carve((size_t)B_ * 1024 * 256 * 4);
    float* f3   = (float*)carve((size_t)B_ * 512  * 256 * 4);
    float* f4   = (float*)carve((size_t)B_ * 256  * 256 * 4);
    float* g3   = (float*)carve((size_t)B_ * 512  * 256 * 4);
    float* cls  = (float*)carve((size_t)B_ * 1024 * 4);
    float* olin = (float*)carve((size_t)B_ * 512  * 4);
    float* sums = (float*)carve(256 * 4);
    float* sqs  = (float*)carve(256 * 4);
    _Float16* wbuf = (_Float16*)carve((size_t)256 * 512 * 2);
    _Float16* bufA = (_Float16*)carve((size_t)262144 * 128 * 2);   // 64 MiB
    _Float16* bufO = (_Float16*)carve((size_t)262144 * 128 * 2);   // 64 MiB

    const int TPB = 256;
    auto nb = [](size_t n, int tpb) { return (unsigned)((n + tpb - 1) / tpb); };

    // ---- MLP runner: for layer l, params at d_in[pbase+3l .. +2] ----
    auto run_mlp = [&](int M, int Kp0, int pbase, const int* dims, int nlayer,
                       float* out32_last) {
        int Kp = Kp0;
        for (int l = 0; l < nlayer; ++l) {
            int Ci = dims[l], Co = dims[l + 1];
            int wtot = Co * Kp;
            k_wconv<<<nb(wtot, TPB), TPB, 0, stream>>>(
                (const float*)d_in[pbase + 3 * l], wbuf, Ci, Kp, wtot);
            // wave strip = 32 rows; block = 4 waves = 128 rows; N strip = 64 cols
            dim3 grid(Co / 64, (M + 32 * GEMM_WAVES - 1) / (32 * GEMM_WAVES));
            k_gemm_wmma<<<grid, 32 * GEMM_WAVES, 0, stream>>>(bufA, wbuf, bufO, M, Kp, Co);
            k_col_stats<<<Co, 256, 0, stream>>>(bufO, M, Co, sums, sqs);
            size_t tot = (size_t)M * Co;
            k_bn_relu<<<nb(tot, TPB), TPB, 0, stream>>>(
                bufO, (const float*)d_in[pbase + 3 * l + 1],
                (const float*)d_in[pbase + 3 * l + 2],
                sums, sqs, M, Co, bufA,
                (l == nlayer - 1) ? out32_last : (float*)nullptr);
            Kp = Co;
        }
    };

    // ---- SA stage runner ----
    auto run_sa = [&](const float* xyz_in, int Nin, const float* feats_in, int Cf,
                      int np, int ns, float radius, int Cpad0, int pbase,
                      const int* dims, float* xyz_out, float* feats_out) {
        size_t lds = ((size_t)4 * Nin + 2 * 1024) * 4;
        k_fps<<<B_, 1024, lds, stream>>>(xyz_in, Nin, np, inds);
        int tc = B_ * np;
        k_gather_xyz<<<nb(tc, TPB), TPB, 0, stream>>>(xyz_in, inds, xyz_out, Nin, np, tc);
        k_ball_query<<<nb(tc, TPB), TPB, 0, stream>>>(
            xyz_in, xyz_out, Nin, np, ns, radius * radius, idx, tc);
        int M = B_ * np * ns;
        size_t gt = (size_t)M * Cpad0;
        k_group<<<nb(gt, TPB), TPB, 0, stream>>>(
            xyz_in, xyz_out, feats_in, idx, bufA, np, ns, Cf, Cpad0,
            1.0f / radius, Nin, gt);
        run_mlp(M, Cpad0, pbase, dims, 3, nullptr);
        int Cl = dims[3];
        int mt = B_ * np * Cl;
        k_max_pool<<<nb(mt, TPB), TPB, 0, stream>>>(bufA, feats_out, ns, Cl, mt);
    };

    // ---- FP stage runner ----
    auto run_fp = [&](const float* xa, int N1, const float* xb, int N2,
                      const float* fa, int C1, const float* fb, int C2,
                      int pbase, const int* dims, float* out32) {
        int tc = B_ * N1;
        k_three_interp<<<nb(tc, TPB), TPB, 0, stream>>>(
            xa, xb, fa, fb, bufA, N1, N2, C1, C2, tc);
        run_mlp(B_ * N1, C1 + C2, pbase, dims, 2, out32);
    };

    // ---- pipeline ----
    const float* pc = (const float*)d_in[0];
    int tt = B_ * N0_;
    k_transpose<<<nb(tt, TPB), TPB, 0, stream>>>(pc, xyz0, N0_, tt);

    static const int sa1d[4] = {3, 64, 64, 128};
    static const int sa2d[4] = {131, 128, 128, 256};
    static const int sa34d[4] = {259, 128, 128, 256};
    static const int fpd[3] = {512, 256, 256};

    run_sa(xyz0, N0_, nullptr, 0,   2048, 64, 0.2f, 32,  1,  sa1d,  xyz1, f1);
    run_sa(xyz1, 2048, f1,    128,  1024, 32, 0.4f, 160, 10, sa2d,  xyz2, f2);
    run_sa(xyz2, 1024, f2,    256,  512,  16, 0.8f, 288, 19, sa34d, xyz3, f3);
    run_sa(xyz3, 512,  f3,    256,  256,  16, 1.2f, 288, 28, sa34d, xyz4, f4);

    run_fp(xyz3, 512,  xyz4, 256, f3, 256, f4, 256, 37, fpd, g3);
    run_fp(xyz2, 1024, xyz3, 512, f2, 256, g3, 256, 43, fpd, nullptr);

    // cls = max over channels of g2 (resident in bufA: 2048 x 256 f16)
    int ct = B_ * 1024;
    k_ch_max<<<nb(ct, TPB), TPB, 0, stream>>>(bufA, cls, 256, ct);
    int lt = B_ * 512;
    k_linear<<<nb(lt, TPB), TPB, 0, stream>>>(cls, (const float*)d_in[49], olin, 1024, 512, lt);
    k_final_bn<<<nb(512, TPB), TPB, 0, stream>>>(
        olin, (const float*)d_in[50], (const float*)d_in[51], (float*)d_out, 512);
}
